// QAPSolverModule_63307817943834
// MI455X (gfx1250) — compile-verified, matrix operand order checked by reading, not run on hardware
//
#include <hip/hip_runtime.h>

// ---- WMMA fragment types (CDNA5 / gfx1250, wave32) ----
typedef __attribute__((ext_vector_type(16))) __bf16 v16bf;
typedef __attribute__((ext_vector_type(8)))  float  v8f;

#define QN 512    // nodes
#define QH 256    // labels
#define QE 4096   // edges
#define QK 8192   // GEMM K = 2*E  ([src | dst] halves)
#define KT32 256  // QK / 32  (K-tiles per WMMA step)
#define NT16 16   // QH / 16
#define MT16 32   // QN / 16

// ---- helpers ----------------------------------------------------------
__device__ __forceinline__ unsigned short f32_bf16(float f) {
  unsigned int u = __float_as_uint(f);
  u += 0x7fffu + ((u >> 16) & 1u);   // round-to-nearest-even
  return (unsigned short)(u >> 16);
}

// Packed fragment storage: one 16x32 (A, MxK) or 32x16 (B, KxN) tile is
// 512 bf16 laid out as [lane 0..31][elem 0..15] so a wave loads its whole
// fragment as one 32-byte v16bf per lane (2x global_load_b128).
// Per ISA 7.12.2 (16-bit operands): lanes 0-15 hold K in {0..7, 16..23},
// lanes 16-31 hold K in {8..15, 24..31}; elem order = K0..7 then K16..23.
__device__ __forceinline__ size_t pack_off(int tile, int kr, int mn) {
  int hi   = (kr >> 3) & 1;
  int lane = mn + (hi << 4);
  int elem = (kr & 7) | ((kr & 16) >> 1);
  return (size_t)tile * 512 + (size_t)lane * 16 + elem;
}

// ---- kernels ----------------------------------------------------------
__global__ void k_zero_u32(unsigned int* p, int n) {
  int i = blockIdx.x * blockDim.x + threadIdx.x;
  if (i < n) p[i] = 0u;
}

// Build incidence matrix S (bf16 1.0 entries) in packed A-fragment layout.
// Row = node, col k = e (src half) or E+e (dst half).
__global__ void k_buildS(const int* __restrict__ edges,
                         unsigned short* __restrict__ Apack) {
  int e = blockIdx.x * blockDim.x + threadIdx.x;
  if (e >= QE) return;
  int s = edges[2 * e + 0];
  int d = edges[2 * e + 1];
  const unsigned short one = 0x3F80;  // bf16(1.0)
  {
    int k = e, kt = k >> 5, kr = k & 31;
    int mt = s >> 4, mr = s & 15;
    Apack[pack_off(mt * KT32 + kt, kr, mr)] = one;
  }
  {
    int k = QE + e, kt = k >> 5, kr = k & 31;
    int mt = d >> 4, mr = d & 15;
    Apack[pack_off(mt * KT32 + kt, kr, mr)] = one;
  }
}

// Per edge: gather row pw[e, lab[src], :] and column pw[e, :, lab[dst]],
// write bf16 into packed B-fragment layout (K x N = 8192 x 256).
__global__ void k_gather(const float* __restrict__ pw,
                         const int* __restrict__ edges,
                         const int* __restrict__ lab,
                         unsigned short* __restrict__ Bpack) {
  int e = blockIdx.x;
  int h = threadIdx.x;                 // label index 0..255
  int s = edges[2 * e + 0];
  int d = edges[2 * e + 1];
  int ls = lab[s];
  int ld = lab[d];
  size_t base = (size_t)e * (QH * QH);
  float colv = pw[base + (size_t)h * QH + ld];  // m_src[e,h] -> k = e
  float rowv = pw[base + (size_t)ls * QH + h];  // m_dst[e,h] -> k = E+e
  int nt = h >> 4, nr = h & 15;
  {
    int k = e, kt = k >> 5, kr = k & 31;
    Bpack[pack_off(kt * NT16 + nt, kr, nr)] = f32_bf16(colv);
  }
  {
    int k = QE + e, kt = k >> 5, kr = k & 31;
    Bpack[pack_off(kt * NT16 + nt, kr, nr)] = f32_bf16(rowv);
  }
}

// cost[512x256] = (u - 1000) + S @ M   via v_wmma_f32_16x16x32_bf16.
// One wave per 16x16 output tile; 256 K-steps of K=32.
__global__ void k_gemm(const unsigned short* __restrict__ Apack,
                       const unsigned short* __restrict__ Bpack,
                       const float* __restrict__ u,
                       float* __restrict__ cost) {
  int wave = threadIdx.x >> 5;
  int lane = threadIdx.x & 31;
  int tile = blockIdx.x * (blockDim.x >> 5) + wave;  // 0..511
  int mt = tile >> 4;   // / NT16
  int nt = tile & 15;

  const v16bf* A = ((const v16bf*)Apack) + (size_t)mt * KT32 * 32 + lane;
  const v16bf* B = ((const v16bf*)Bpack) + (size_t)nt * 32 + lane;

  v8f c = {0.f, 0.f, 0.f, 0.f, 0.f, 0.f, 0.f, 0.f};
#pragma unroll 8
  for (int kt = 0; kt < KT32; ++kt) {
    v16bf a = A[(size_t)kt * 32];
    v16bf b = B[(size_t)kt * NT16 * 32];
    c = __builtin_amdgcn_wmma_f32_16x16x32_bf16(
        /*neg_a=*/false, a, /*neg_b=*/false, b,
        /*c_mod=*/(short)0, c, /*reuse_a=*/false, /*reuse_b=*/false);
  }

  // C/D layout: VGPR v, lanes 0-15 -> M=v, lanes 16-31 -> M=v+8; N=lane&15.
  int n    = nt * 16 + (lane & 15);
  int mrow = mt * 16 + (lane >> 4) * 8;
#pragma unroll
  for (int v = 0; v < 8; ++v) {
    int m = mrow + v;
    cost[(size_t)m * QH + n] = u[(size_t)m * QH + n] - 1000.0f + c[v];
  }
}

// First-occurrence argmin over H=256 per node (jnp.argmin tie-break).
__global__ void k_argmin(const float* __restrict__ mat, int* __restrict__ lab) {
  __shared__ float sv[256];
  __shared__ int   si[256];
  int n = blockIdx.x, h = threadIdx.x;
  sv[h] = mat[(size_t)n * QH + h];
  si[h] = h;
  __syncthreads();
  for (int off = 128; off > 0; off >>= 1) {
    if (h < off) {
      float a = sv[h], b = sv[h + off];
      int ia = si[h], ib = si[h + off];
      if (b < a || (b == a && ib < ia)) { sv[h] = b; si[h] = ib; }
    }
    __syncthreads();
  }
  if (h == 0) lab[n] = si[0];
}

__global__ void k_onehot(const int* __restrict__ lab, float* __restrict__ out) {
  int n = blockIdx.x, h = threadIdx.x;
  out[(size_t)n * QH + h] = (lab[n] == h) ? 1.0f : 0.0f;
}

// ---- launch -----------------------------------------------------------
extern "C" void kernel_launch(void* const* d_in, const int* in_sizes, int n_in,
                              void* d_out, int out_size, void* d_ws, size_t ws_size,
                              hipStream_t stream) {
  (void)in_sizes; (void)n_in; (void)out_size; (void)ws_size;

  const float* u     = (const float*)d_in[0];  // [512,256] f32
  const float* pw    = (const float*)d_in[1];  // [4096,256,256] f32
  const int*   edges = (const int*)d_in[2];    // [4096,2] i32
  float*       out   = (float*)d_out;          // [512,256] f32 one-hot

  char* ws = (char*)d_ws;
  unsigned short* Apack = (unsigned short*)(ws);                         // 8 MiB
  unsigned short* Bpack = (unsigned short*)(ws + (8u << 20));            // 4 MiB
  float*          cost  = (float*)(ws + (12u << 20));                    // 512 KiB
  int*            labA  = (int*)(ws + (12u << 20) + (512u << 10));       // 2 KiB
  int*            labB  = labA + QN;                                     // 2 KiB

  // Build packed incidence matrix S (iteration-invariant).
  const int a_u32 = (QN * QK) / 2;  // bf16 elems / 2
  k_zero_u32<<<(a_u32 + 255) / 256, 256, 0, stream>>>((unsigned int*)Apack, a_u32);
  k_buildS<<<QE / 256, 256, 0, stream>>>(edges, Apack);

  // lab0 = argmin(u) (the -1000 shift doesn't change argmin).
  k_argmin<<<QN, 256, 0, stream>>>(u, labA);

  int* cur = labA;
  int* nxt = labB;
  for (int it = 0; it < 10; ++it) {
    k_gather<<<QE, 256, 0, stream>>>(pw, edges, cur, Bpack);
    k_gemm<<<MT16 * NT16 / 8, 256, 0, stream>>>(Apack, Bpack, u, cost);
    k_argmin<<<QN, 256, 0, stream>>>(cost, nxt);
    int* t = cur; cur = nxt; nxt = t;
  }

  k_onehot<<<QN, 256, 0, stream>>>(cur, out);
}
